// Attention_32203664785557
// MI455X (gfx1250) — matrix operand order repaired and checked by reading, not compiled, not run on hardware
//
#include <hip/hip_runtime.h>
#include <hip/hip_bf16.h>
#include <math.h>

// ---------------- problem constants ----------------
#define BATCH   8
#define SEQ     1024
#define DIM     768
#define HEADS   12
#define HD      64
#define NQKV    (3 * DIM)          // 2304
#define ATT_SCALE 0.125f           // 64^-0.5
#define LOG2E   1.44269504088896f

typedef __bf16 bf16;
typedef __attribute__((ext_vector_type(16))) __bf16 bf16x16;
typedef __attribute__((ext_vector_type(8)))  float  f32x8;

union Frag {
    bf16x16 v;
    uint4   q[2];
};

// float -> bf16, round-to-nearest-even
__device__ __forceinline__ bf16 f2bf(float f) {
    unsigned u = __float_as_uint(f);
    u += 0x7FFFu + ((u >> 16) & 1u);
    unsigned short h = (unsigned short)(u >> 16);
    bf16 r;
    __builtin_memcpy(&r, &h, 2);
    return r;
}

// A-fragment (16x32 bf16, MxK): lane holds row m = lane%16, lg = lane/16.
// element e -> k = (e/8)*16 + lg*8 + (e%8)  => two contiguous 16B loads.
__device__ __forceinline__ bf16x16 load_a_frag(const bf16* row_base, int lg) {
    Frag f;
    f.q[0] = *(const uint4*)(row_base + lg * 8);
    f.q[1] = *(const uint4*)(row_base + 16 + lg * 8);
    return f.v;
}

// B-fragment (32x16 bf16, KxN) from B^T storage (n-major rows, k contiguous):
// lane holds col n = lane%16; element e -> k = (e/8)*16 + lg*8 + (e%8) => two 16B loads.
__device__ __forceinline__ bf16x16 load_b_frag(const bf16* colrow_base, int lg) {
    Frag f;
    f.q[0] = *(const uint4*)(colrow_base + lg * 8);
    f.q[1] = *(const uint4*)(colrow_base + 16 + lg * 8);
    return f.v;
}

__device__ __forceinline__ f32x8 wmma_bf16(bf16x16 a, bf16x16 b, f32x8 c) {
    return __builtin_amdgcn_wmma_f32_16x16x32_bf16(
        /*neg_a=*/false, a, /*neg_b=*/false, b,
        /*c_mod=*/(short)0, c, /*reuse_a=*/false, /*reuse_b=*/false);
}

// ---------------- conversion kernels ----------------
__global__ void cvt_kernel(const float* __restrict__ src, bf16* __restrict__ dst, int n) {
    int i = blockIdx.x * blockDim.x + threadIdx.x;
    if (i < n) dst[i] = f2bf(src[i]);
}

// src: K x N row-major fp32 -> dst: N x K row-major bf16 (transposed)
__global__ void cvt_t_kernel(const float* __restrict__ src, bf16* __restrict__ dst, int K, int N) {
    int i = blockIdx.x * blockDim.x + threadIdx.x;
    if (i < K * N) {
        int k = i / N, n = i % N;
        dst[(size_t)n * K + k] = f2bf(src[i]);
    }
}

// ---------------- GEMM core (wave tile 32M x 64N, double-buffered) ----------
// acc[r][t] over r in {0,1} (row sub-tiles of 16) and t in {0..3} (col tiles).
#define GEMM_CORE(A_, BT_, ROW0_, COL0_, ACC_)                                  \
    const bf16* arow0 = (A_) + (size_t)((ROW0_) + lr) * DIM;                    \
    const bf16* arow1 = (A_) + (size_t)((ROW0_) + 16 + lr) * DIM;               \
    const bf16* brow0 = (BT_) + (size_t)((COL0_) + 0 * 16 + lr) * DIM;          \
    const bf16* brow1 = (BT_) + (size_t)((COL0_) + 1 * 16 + lr) * DIM;          \
    const bf16* brow2 = (BT_) + (size_t)((COL0_) + 2 * 16 + lr) * DIM;          \
    const bf16* brow3 = (BT_) + (size_t)((COL0_) + 3 * 16 + lr) * DIM;          \
    bf16x16 ca0 = load_a_frag(arow0, lg), ca1 = load_a_frag(arow1, lg);         \
    bf16x16 cb0 = load_b_frag(brow0, lg), cb1 = load_b_frag(brow1, lg);         \
    bf16x16 cb2 = load_b_frag(brow2, lg), cb3 = load_b_frag(brow3, lg);         \
    for (int kk = 32; kk < DIM; kk += 32) {                                     \
        /* issue next-tile loads before current WMMAs: overlap mem/matrix */    \
        bf16x16 na0 = load_a_frag(arow0 + kk, lg);                              \
        bf16x16 na1 = load_a_frag(arow1 + kk, lg);                              \
        bf16x16 nb0 = load_b_frag(brow0 + kk, lg);                              \
        bf16x16 nb1 = load_b_frag(brow1 + kk, lg);                              \
        bf16x16 nb2 = load_b_frag(brow2 + kk, lg);                              \
        bf16x16 nb3 = load_b_frag(brow3 + kk, lg);                              \
        if (kk + 32 < DIM) {                                                    \
            /* locality 3 -> near (WGP) scope prefetch for the k+2 tile */      \
            __builtin_prefetch(arow0 + kk + 32, 0, 3);                          \
            __builtin_prefetch(brow0 + kk + 32, 0, 3);                          \
            __builtin_prefetch(brow2 + kk + 32, 0, 3);                          \
        }                                                                       \
        ACC_[0][0] = wmma_bf16(ca0, cb0, ACC_[0][0]);                           \
        ACC_[0][1] = wmma_bf16(ca0, cb1, ACC_[0][1]);                           \
        ACC_[0][2] = wmma_bf16(ca0, cb2, ACC_[0][2]);                           \
        ACC_[0][3] = wmma_bf16(ca0, cb3, ACC_[0][3]);                           \
        ACC_[1][0] = wmma_bf16(ca1, cb0, ACC_[1][0]);                           \
        ACC_[1][1] = wmma_bf16(ca1, cb1, ACC_[1][1]);                           \
        ACC_[1][2] = wmma_bf16(ca1, cb2, ACC_[1][2]);                           \
        ACC_[1][3] = wmma_bf16(ca1, cb3, ACC_[1][3]);                           \
        ca0 = na0; ca1 = na1;                                                   \
        cb0 = nb0; cb1 = nb1; cb2 = nb2; cb3 = nb3;                             \
    }                                                                           \
    ACC_[0][0] = wmma_bf16(ca0, cb0, ACC_[0][0]);                               \
    ACC_[0][1] = wmma_bf16(ca0, cb1, ACC_[0][1]);                               \
    ACC_[0][2] = wmma_bf16(ca0, cb2, ACC_[0][2]);                               \
    ACC_[0][3] = wmma_bf16(ca0, cb3, ACC_[0][3]);                               \
    ACC_[1][0] = wmma_bf16(ca1, cb0, ACC_[1][0]);                               \
    ACC_[1][1] = wmma_bf16(ca1, cb1, ACC_[1][1]);                               \
    ACC_[1][2] = wmma_bf16(ca1, cb2, ACC_[1][2]);                               \
    ACC_[1][3] = wmma_bf16(ca1, cb3, ACC_[1][3]);

// ---------------- QKV GEMM ----------------
// C[8192, 2304] = Xh[8192,768] @ Wqkv ; B given transposed (Bt[2304][768]).
// Scatter: q*SCALE -> Q[B,H,N,D], k -> K[B,H,N,D], v -> Vt[B,H,D,N].
__global__ void __launch_bounds__(128) qkv_gemm_kernel(
    const bf16* __restrict__ A, const bf16* __restrict__ Bt,
    bf16* __restrict__ Qo, bf16* __restrict__ Ko, bf16* __restrict__ Vt) {
    const int lane = threadIdx.x & 31;
    const int wave = threadIdx.x >> 5;
    const int lg = lane >> 4, lr = lane & 15;
    const int row0 = blockIdx.x * 128 + wave * 32;
    const int col0 = blockIdx.y * 64;

    f32x8 acc[2][4] = {};
    GEMM_CORE(A, Bt, row0, col0, acc)

#pragma unroll
    for (int r = 0; r < 2; ++r) {
#pragma unroll
        for (int t = 0; t < 4; ++t) {
            int c = col0 + t * 16 + lr;
            int which = c / DIM;
            int rr = c % DIM;
            int h = rr / HD, d = rr % HD;
#pragma unroll
            for (int i = 0; i < 8; ++i) {
                int row = row0 + r * 16 + lg * 8 + i;
                int b = row / SEQ, n = row % SEQ;
                float v = acc[r][t][i];
                if (which == 0)
                    Qo[(((size_t)b * HEADS + h) * SEQ + n) * HD + d] = f2bf(v * ATT_SCALE);
                else if (which == 1)
                    Ko[(((size_t)b * HEADS + h) * SEQ + n) * HD + d] = f2bf(v);
                else
                    Vt[(((size_t)b * HEADS + h) * HD + d) * SEQ + n] = f2bf(v);
            }
        }
    }
}

// ---------------- flash attention ----------------
// One wave = 16 query rows; block = 4 waves = 64 rows of one (b,h).
// Online softmax over key blocks of 32; O accumulated in 4 f32 C-fragments.
// K fragments are double-buffered across column blocks so their load latency
// hides under the softmax (shuffle/exp) and P.V WMMA work.
__global__ void __launch_bounds__(128) attn_kernel(
    const bf16* __restrict__ Q, const bf16* __restrict__ K,
    const bf16* __restrict__ Vt, bf16* __restrict__ O) {
    __shared__ __attribute__((aligned(16))) bf16 pls[4][16][32];

    const int lane = threadIdx.x & 31;
    const int wave = threadIdx.x >> 5;
    const int lg = lane >> 4, lr = lane & 15;
    const int bh = blockIdx.x >> 4;              // 0..95
    const int row0 = (blockIdx.x & 15) * 64 + wave * 16;
    const int b = bh / HEADS, h = bh % HEADS;

    const bf16* Qb = Q + (size_t)bh * SEQ * HD;
    const bf16* Kb = K + (size_t)bh * SEQ * HD;
    const bf16* Vb = Vt + (size_t)bh * HD * SEQ;

    // Q fragments for this wave's 16 rows (head-dim chunks 0..31, 32..63)
    bf16x16 qf0 = load_a_frag(Qb + (size_t)(row0 + lr) * HD, lg);
    bf16x16 qf1 = load_a_frag(Qb + (size_t)(row0 + lr) * HD + 32, lg);

    f32x8 o[4] = {};
    float m[8], l[8];
#pragma unroll
    for (int i = 0; i < 8; ++i) { m[i] = -1e30f; l[i] = 0.0f; }

    // preload K fragments for column block 0
    bf16x16 ck00 = load_b_frag(Kb + (size_t)lr * HD, lg);
    bf16x16 ck01 = load_b_frag(Kb + (size_t)lr * HD + 32, lg);
    bf16x16 ck10 = load_b_frag(Kb + (size_t)(16 + lr) * HD, lg);
    bf16x16 ck11 = load_b_frag(Kb + (size_t)(16 + lr) * HD + 32, lg);

    for (int cb = 0; cb < SEQ; cb += 32) {
        // V fragments for this block (consumed at the end of the iteration)
        bf16x16 vf[4];
#pragma unroll
        for (int t = 0; t < 4; ++t)
            vf[t] = load_b_frag(Vb + (size_t)(t * 16 + lr) * SEQ + cb, lg);

        f32x8 s0 = {}, s1 = {};
        s0 = wmma_bf16(qf0, ck00, s0);
        s0 = wmma_bf16(qf1, ck01, s0);
        s1 = wmma_bf16(qf0, ck10, s1);
        s1 = wmma_bf16(qf1, ck11, s1);

        // issue next block's K loads now; latency hides under softmax + P.V
        const int nb = (cb + 32) & (SEQ - 1);    // wrap keeps loads in-bounds
        bf16x16 nk00 = load_b_frag(Kb + (size_t)(nb + lr) * HD, lg);
        bf16x16 nk01 = load_b_frag(Kb + (size_t)(nb + lr) * HD + 32, lg);
        bf16x16 nk10 = load_b_frag(Kb + (size_t)(nb + 16 + lr) * HD, lg);
        bf16x16 nk11 = load_b_frag(Kb + (size_t)(nb + 16 + lr) * HD + 32, lg);

        // online softmax; row m=lg*8+i lives across the 16 lanes sharing lg
#pragma unroll
        for (int i = 0; i < 8; ++i) {
            float a0 = s0[i], a1 = s1[i];
            float mx = fmaxf(a0, a1);
            mx = fmaxf(mx, __shfl_xor(mx, 1, 32));
            mx = fmaxf(mx, __shfl_xor(mx, 2, 32));
            mx = fmaxf(mx, __shfl_xor(mx, 4, 32));
            mx = fmaxf(mx, __shfl_xor(mx, 8, 32));
            float mnew = fmaxf(m[i], mx);
            float p0 = exp2f((a0 - mnew) * LOG2E);
            float p1 = exp2f((a1 - mnew) * LOG2E);
            float corr = exp2f((m[i] - mnew) * LOG2E);
            float rs = p0 + p1;
            rs += __shfl_xor(rs, 1, 32);
            rs += __shfl_xor(rs, 2, 32);
            rs += __shfl_xor(rs, 4, 32);
            rs += __shfl_xor(rs, 8, 32);
            l[i] = l[i] * corr + rs;
            m[i] = mnew;
#pragma unroll
            for (int t = 0; t < 4; ++t) o[t][i] *= corr;
            pls[wave][lg * 8 + i][lr]      = f2bf(p0);
            pls[wave][lg * 8 + i][16 + lr] = f2bf(p1);
        }
        // same-wave LDS is in-order; wait out our own ds stores before reload
        asm volatile("s_wait_dscnt 0" ::: "memory");
        bf16x16 pf = load_a_frag(&pls[wave][lr][0], lg);
#pragma unroll
        for (int t = 0; t < 4; ++t)
            o[t] = wmma_bf16(pf, vf[t], o[t]);

        ck00 = nk00; ck01 = nk01; ck10 = nk10; ck11 = nk11;
    }
    // epilogue -> O[b][n][h*64+d] bf16
#pragma unroll
    for (int i = 0; i < 8; ++i) {
        float inv = 1.0f / l[i];
        int n = row0 + lg * 8 + i;
        size_t base = ((size_t)b * SEQ + n) * DIM + h * HD;
#pragma unroll
        for (int t = 0; t < 4; ++t)
            O[base + t * 16 + lr] = f2bf(o[t][i] * inv);
    }
}

// ---------------- output projection ----------------
// out[8192,768] = Ob[8192,768] @ Wproj + bias, Wproj given transposed.
__global__ void __launch_bounds__(128) proj_gemm_kernel(
    const bf16* __restrict__ A, const bf16* __restrict__ Bt,
    const float* __restrict__ bias, float* __restrict__ Cout) {
    const int lane = threadIdx.x & 31;
    const int wave = threadIdx.x >> 5;
    const int lg = lane >> 4, lr = lane & 15;
    const int row0 = blockIdx.x * 128 + wave * 32;
    const int col0 = blockIdx.y * 64;

    f32x8 acc[2][4] = {};
    GEMM_CORE(A, Bt, row0, col0, acc)

#pragma unroll
    for (int r = 0; r < 2; ++r) {
#pragma unroll
        for (int t = 0; t < 4; ++t) {
            int c = col0 + t * 16 + lr;
            float bv = bias[c];
#pragma unroll
            for (int i = 0; i < 8; ++i) {
                int row = row0 + r * 16 + lg * 8 + i;
                Cout[(size_t)row * DIM + c] = acc[r][t][i] + bv;
            }
        }
    }
}

// ---------------- launch ----------------
extern "C" void kernel_launch(void* const* d_in, const int* in_sizes, int n_in,
                              void* d_out, int out_size, void* d_ws, size_t ws_size,
                              hipStream_t stream) {
    const float* x      = (const float*)d_in[0];
    const float* w_qkv  = (const float*)d_in[1];
    const float* w_proj = (const float*)d_in[2];
    const float* b_proj = (const float*)d_in[3];
    float* out = (float*)d_out;

    const size_t NROW = (size_t)BATCH * SEQ;      // 8192
    char* ws = (char*)d_ws;
    size_t off = 0;
    auto alloc = [&](size_t bytes) {
        void* p = ws + off;
        off += (bytes + 255) & ~(size_t)255;
        return p;
    };
    bf16* xh      = (bf16*)alloc(NROW * DIM * sizeof(bf16));           // 12 MB
    bf16* wqkv_t  = (bf16*)alloc((size_t)NQKV * DIM * sizeof(bf16));   // 3.5 MB
    bf16* wproj_t = (bf16*)alloc((size_t)DIM * DIM * sizeof(bf16));    // 1.2 MB
    bf16* Qb      = (bf16*)alloc(NROW * DIM * sizeof(bf16));           // [B,H,N,D]
    bf16* Kb      = (bf16*)alloc(NROW * DIM * sizeof(bf16));
    bf16* Vt      = (bf16*)alloc(NROW * DIM * sizeof(bf16));           // [B,H,D,N]
    bf16* Ob      = (bf16*)alloc(NROW * DIM * sizeof(bf16));
    (void)ws_size; (void)in_sizes; (void)n_in; (void)out_size;         // ~68 MB total

    // 1) convert inputs to bf16 (weights transposed for contiguous B-fragments)
    {
        int n = (int)(NROW * DIM);
        cvt_kernel<<<(n + 255) / 256, 256, 0, stream>>>(x, xh, n);
        int nw = DIM * NQKV;
        cvt_t_kernel<<<(nw + 255) / 256, 256, 0, stream>>>(w_qkv, wqkv_t, DIM, NQKV);
        int np = DIM * DIM;
        cvt_t_kernel<<<(np + 255) / 256, 256, 0, stream>>>(w_proj, wproj_t, DIM, DIM);
    }
    // 2) QKV GEMM + scatter to Q (scaled), K, V^T   (block tile 128x64)
    qkv_gemm_kernel<<<dim3(NROW / 128, NQKV / 64), 128, 0, stream>>>(xh, wqkv_t, Qb, Kb, Vt);
    // 3) flash attention: 96 (b,h) x 16 row-blocks
    attn_kernel<<<BATCH * HEADS * (SEQ / 64), 128, 0, stream>>>(Qb, Kb, Vt, Ob);
    // 4) projection + bias -> fp32 out   (block tile 128x64)
    proj_gemm_kernel<<<dim3(NROW / 128, DIM / 64), 128, 0, stream>>>(Ob, wproj_t, b_proj, out);
}
